// AttentionWithFastKANTransform_53953379172438
// MI455X (gfx1250) — matrix operand order repaired and checked by reading, not compiled, not run on hardware
//
#include <hip/hip_runtime.h>
#include <hip/hip_bf16.h>

// ---------------------------------------------------------------------------
// Problem constants (from reference)
// ---------------------------------------------------------------------------
#define D_IN      256
#define D_OUT     256
#define K_SPLINE  2048          // 256 features * 8 grids
#define B_ROWS    128
#define S_ROWS    8192
#define HEADS     8
#define HDIM      32
#define INV_DENOM 1.75f         // (NUM_GRIDS-1)/(GRID_MAX-GRID_MIN) = 7/4
#define NORM      0.17677669529663689f   // 1/sqrt(32)

typedef __attribute__((ext_vector_type(16))) __bf16 v16bf;
typedef __attribute__((ext_vector_type(8)))  float  v8f;
typedef __attribute__((ext_vector_type(2)))  float  v2f;
typedef __attribute__((ext_vector_type(2)))  __bf16 v2bf;

static __device__ __forceinline__ v8f wmma_bf16(v16bf a, v16bf b, v8f c) {
  // D = A(16x32 bf16) * B(32x16 bf16) + C(16x16 f32)
  return __builtin_amdgcn_wmma_f32_16x16x32_bf16(false, a, false, b,
                                                 (short)0, c, false, false);
}

// Scalar f32 -> bf16 via the compiler's native conversion (hw cvt if present).
static __device__ __forceinline__ unsigned short f2bf(float f) {
  return __builtin_bit_cast(unsigned short, (__bf16)f);
}
// Packed pair conversion: lets the backend select v_cvt_pk_bf16_f32.
static __device__ __forceinline__ unsigned pack2bf(float lo, float hi) {
  v2f f; f[0] = lo; f[1] = hi;
  return __builtin_bit_cast(unsigned, __builtin_convertvector(f, v2bf));
}

// A fragment (16x32 bf16, ISA layout): two aligned 16B LDS loads.
// Rows padded to 40 halfs (80B) so every chunk is 16B aligned.
static __device__ __forceinline__ v16bf frag_a(const unsigned short* At, int lane) {
  const int r  = lane & 15;
  const int kb = (lane < 16) ? 0 : 8;
  union { uint4 q[2]; v16bf v; } u;
  u.q[0] = *(const uint4*)(At + r * 40 + kb);        // halves 0..7  -> K kb..kb+7
  u.q[1] = *(const uint4*)(At + r * 40 + 16 + kb);   // halves 8..15 -> K 16+kb..
  return u.v;
}

// B fragment (32x16 bf16) from a TRANSPOSED tile BtT[col][k] (40-half rows):
// lane owns column n, K = koff..koff+7 and 16+koff..16+koff+7 -> two aligned 16B loads.
static __device__ __forceinline__ v16bf frag_bT(const unsigned short* BtT,
                                                int lane, int col0) {
  const int n    = lane & 15;
  const int koff = (lane < 16) ? 0 : 8;
  union { uint4 q[2]; v16bf v; } u;
  u.q[0] = *(const uint4*)(BtT + (col0 + n) * 40 + koff);
  u.q[1] = *(const uint4*)(BtT + (col0 + n) * 40 + koff + 16);
  return u.v;
}

// ---------------------------------------------------------------------------
// Fused FastKAN layer:  out[N,256] = [basis(X) | silu(X)] @ [[SW],[BW]] + BB
// Grid: (N/16, 256/64), block 128 = 4 waves, each wave one 16x16 WMMA tile.
// RBF recurrence: with c=(7/4)^2, dg=4/7 => c*dg^2 == 1 exactly, so
//   f_{g+1} = f_g * r_g,  r_{g+1} = r_g * e^-2   (2 exps for 8 gaussians).
// ---------------------------------------------------------------------------
__launch_bounds__(128)
__global__ void fastkan_kernel(const float* __restrict__ X,
                               const float* __restrict__ SW,
                               const float* __restrict__ BW,
                               const float* __restrict__ BB,
                               float scale, int act,
                               float* __restrict__ outF,
                               unsigned short* __restrict__ outB) {
  __shared__ float Xs[16][D_IN];                        // 16 KB input rows
  __shared__ __align__(16) unsigned short At[16][40];   // A tile (bf16)
  __shared__ __align__(16) unsigned short BtT[64][40];  // B tile transposed

  const int t = threadIdx.x, lane = t & 31, wave = t >> 5;
  const int rbase = blockIdx.x * 16, cbase = blockIdx.y * 64;
  const int wcol = wave * 16;

  for (int i = 0; i < 32; ++i) {
    int idx = i * 128 + t;
    int r = idx >> 8, c = idx & 255;
    Xs[r][c] = X[(size_t)(rbase + r) * D_IN + c];
  }
  __syncthreads();

  v8f acc;
  #pragma unroll
  for (int j = 0; j < 8; ++j) acc[j] = 0.0f;

  const float C2 = INV_DENOM * INV_DENOM;   // 3.0625 (exact)
  const float Q  = 0.13533528323661270f;    // exp(-2)

  // ---------------- spline phase (K = 0..2047) ----------------
  for (int kc = 0; kc < K_SPLINE; kc += 32) {
    // A tile: 16 rows x (4 features x 8 grids). Threads 0..63: one (row,feat).
    if (t < 64) {
      int r  = t >> 2;
      int jf = (kc >> 3) + (t & 3);
      float x  = Xs[r][jf];
      float d0 = x + 2.0f;                         // x - grid[0]
      float f  = __expf(-C2 * d0 * d0);            // gaussian at grid[0]
      float rr = __expf(3.5f * d0 - 1.0f);         // 2*C2*DG == 7/2 exactly
      float fv[8];
      #pragma unroll
      for (int g = 0; g < 8; ++g) { fv[g] = f; f *= rr; rr *= Q; }
      uint4 pk;
      pk.x = pack2bf(fv[0], fv[1]);
      pk.y = pack2bf(fv[2], fv[3]);
      pk.z = pack2bf(fv[4], fv[5]);
      pk.w = pack2bf(fv[6], fv[7]);
      *(uint4*)&At[r][(t & 3) * 8] = pk;           // one ds_store_b128
    }
    // B tile: 32k x 64c from SW, coalesced read, transposed LDS store.
    for (int i = 0; i < 16; ++i) {
      int idx = i * 128 + t;
      int kk = idx >> 6, cc = idx & 63;
      const float* src = SW + (size_t)(kc + kk) * D_OUT + cbase + cc;
      BtT[cc][kk] = f2bf(*src);
      if (kc + 32 < K_SPLINE) __builtin_prefetch(src + 32 * D_OUT, 0, 0);
    }
    __syncthreads();
    acc = wmma_bf16(frag_a(&At[0][0], lane), frag_bT(&BtT[0][0], lane, wcol), acc);
    __syncthreads();
  }

  // ---------------- silu-base phase (K = 2048..2303) ----------------
  for (int kc = 0; kc < D_IN; kc += 32) {
    for (int i = 0; i < 4; ++i) {
      int idx = i * 128 + t;
      int r = idx >> 5, kk = idx & 31;
      float x = Xs[r][kc + kk];
      At[r][kk] = f2bf(x * __builtin_amdgcn_rcpf(1.0f + __expf(-x)));
    }
    for (int i = 0; i < 16; ++i) {
      int idx = i * 128 + t;
      int kk = idx >> 6, cc = idx & 63;
      BtT[cc][kk] = f2bf(BW[(size_t)(kc + kk) * D_OUT + cbase + cc]);
    }
    __syncthreads();
    acc = wmma_bf16(frag_a(&At[0][0], lane), frag_bT(&BtT[0][0], lane, wcol), acc);
    __syncthreads();
  }

  // ---- epilogue: bias, scale, optional sigmoid; f32 and/or bf16 stores ----
  int n  = lane & 15;
  int mb = (lane < 16) ? 0 : 8;                    // ISA C layout
  #pragma unroll
  for (int j = 0; j < 8; ++j) {
    int row = rbase + mb + j;
    int col = cbase + wcol + n;
    float v = (acc[j] + BB[col]) * scale;
    if (act == 1) v = __builtin_amdgcn_rcpf(1.0f + __expf(-v));
    if (outF) outF[(size_t)row * D_OUT + col] = v;
    if (outB) outB[(size_t)row * D_OUT + col] = f2bf(v);
  }
}

// ---------------------------------------------------------------------------
// Flash attention over 8 heads of dim 32, S = 8192 keys, 16 queries/block.
// 4 waves split S into interleaved 32-key chunks (online softmax per wave),
// V chunks staged with async global->LDS DMA; partials merged through LDS;
// gate g[] fused into the epilogue.
// ---------------------------------------------------------------------------
__launch_bounds__(128)
__global__ void attention_kernel(const unsigned short* __restrict__ wq16,
                                 const unsigned short* __restrict__ wk16,
                                 const unsigned short* __restrict__ wv16,
                                 const float* __restrict__ gbuf,
                                 float* __restrict__ o2) {
  __shared__ __align__(16) unsigned short Ps[4][16][32];   // P tiles (bf16)
  __shared__ __align__(16) unsigned short Vs[4][32][32];   // V chunks (bf16)
  __shared__ float accS[4][16][32];                        // merge buffers
  __shared__ float mS[4][16];
  __shared__ float lS[4][16];

  const int t = threadIdx.x, lane = t & 31, wave = t >> 5;
  const int head = blockIdx.x, qbase = blockIdx.y * 16;

  // Q fragment (A layout) straight from global: two b128 loads per lane.
  v16bf aq;
  {
    int q  = qbase + (lane & 15);
    int kb = (lane < 16) ? 0 : 8;
    const unsigned short* p = wq16 + (size_t)q * 256 + head * HDIM;
    union { uint4 qq[2]; v16bf v; } u;
    u.qq[0] = *(const uint4*)(p + kb);
    u.qq[1] = *(const uint4*)(p + 16 + kb);
    aq = u.v;
  }

  v8f acc0, acc1;
  float mrun[8], lrun[8];
  #pragma unroll
  for (int j = 0; j < 8; ++j) { acc0[j] = 0.f; acc1[j] = 0.f; mrun[j] = -1e30f; lrun[j] = 0.f; }

  for (int ch = wave; ch < S_ROWS / 32; ch += 4) {
    const int sc = ch * 32;

    // Async-DMA this wave's V row (64B) into LDS: global_load_async_to_lds.
    // INST_OFFSET advances both the LDS and global addresses (ISA 10.x).
    {
      unsigned ldsa = (unsigned)(size_t)&Vs[wave][lane][0];   // LDS offset
      const unsigned short* ga = wv16 + (size_t)(sc + lane) * 256 + head * HDIM;
      asm volatile(
        "global_load_async_to_lds_b128 %0, %1, off\n\t"
        "global_load_async_to_lds_b128 %0, %1, off offset:16\n\t"
        "global_load_async_to_lds_b128 %0, %1, off offset:32\n\t"
        "global_load_async_to_lds_b128 %0, %1, off offset:48"
        :: "v"(ldsa), "v"(ga) : "memory");
    }

    // K^T B-fragments straight from global: two 8-bf16 chunks (koff, 16+koff) per lane.
    v16bf bk0, bk1;
    {
      int s0   = sc + (lane & 15);
      int koff = (lane < 16) ? 0 : 8;
      const uint4* p0 = (const uint4*)(wk16 + (size_t)s0 * 256 + head * HDIM + koff);
      const uint4* p1 = (const uint4*)(wk16 + (size_t)(s0 + 16) * 256 + head * HDIM + koff);
      union { uint4 q[2]; v16bf v; } u0, u1;
      u0.q[0] = p0[0]; u0.q[1] = p0[2];
      u1.q[0] = p1[0]; u1.q[1] = p1[2];
      bk0 = u0.v; bk1 = u1.v;
    }

    v8f c0, c1;
    #pragma unroll
    for (int j = 0; j < 8; ++j) { c0[j] = 0.f; c1[j] = 0.f; }
    c0 = wmma_bf16(aq, bk0, c0);        // scores, keys sc..sc+15
    c1 = wmma_bf16(aq, bk1, c1);        // scores, keys sc+16..sc+31

    // Online softmax (row reductions across the 16 lanes holding each row).
    #pragma unroll
    for (int j = 0; j < 8; ++j) {
      float v = fmaxf(c0[j], c1[j]);
      for (int off = 1; off < 16; off <<= 1) v = fmaxf(v, __shfl_xor(v, off, 32));
      float mnew  = fmaxf(mrun[j], v);
      float alpha = __expf(mrun[j] - mnew);
      float e0 = __expf(c0[j] - mnew), e1 = __expf(c1[j] - mnew);
      float ps = e0 + e1;
      for (int off = 1; off < 16; off <<= 1) ps += __shfl_xor(ps, off, 32);
      lrun[j] = lrun[j] * alpha + ps;
      mrun[j] = mnew;
      acc0[j] *= alpha; acc1[j] *= alpha;
      int row = j + ((lane < 16) ? 0 : 8);
      Ps[wave][row][lane & 15]        = f2bf(e0);   // b16 stores (hw cvt)
      Ps[wave][row][16 + (lane & 15)] = f2bf(e1);
    }

    // P fragment: rows are 64B here, so chunks are naturally 16B-aligned.
    v16bf ap;
    {
      int r  = lane & 15;
      int kb = (lane < 16) ? 0 : 16;
      union { uint4 q[2]; v16bf v; } u;
      u.q[0] = *(const uint4*)(&Ps[wave][r][0] + kb);
      u.q[1] = *(const uint4*)(&Ps[wave][r][0] + 8 + kb);
      ap = u.v;
    }

    // Wait for the async V copy, then gather the (strided) V B-fragments.
    asm volatile("s_wait_asynccnt 0" ::: "memory");
    v16bf bv0, bv1;
    {
      int n    = lane & 15;
      int koff = (lane < 16) ? 0 : 16;
      #pragma unroll
      for (int h = 0; h < 16; ++h) {
        bv0[h] = __builtin_bit_cast(__bf16, Vs[wave][koff + h][n]);
        bv1[h] = __builtin_bit_cast(__bf16, Vs[wave][koff + h][16 + n]);
      }
    }
    acc0 = wmma_bf16(ap, bv0, acc0);    // out dims 0..15
    acc1 = wmma_bf16(ap, bv1, acc1);    // out dims 16..31
  }

  // Publish per-wave partials.
  #pragma unroll
  for (int j = 0; j < 8; ++j) {
    int row = j + ((lane < 16) ? 0 : 8);
    accS[wave][row][lane & 15]        = acc0[j];
    accS[wave][row][16 + (lane & 15)] = acc1[j];
  }
  if ((lane & 15) == 0) {
    int half = lane >> 4;
    for (int j = 0; j < 8; ++j) {
      int row = j + half * 8;
      mS[wave][row] = mrun[j];
      lS[wave][row] = lrun[j];
    }
  }
  __syncthreads();

  // Merge the 4 split-S partials with max rescaling; fuse gate multiply.
  for (int i = 0; i < 4; ++i) {
    int idx = i * 128 + t;
    int r = idx >> 5, d = idx & 31;
    float M = fmaxf(fmaxf(mS[0][r], mS[1][r]), fmaxf(mS[2][r], mS[3][r]));
    float num = 0.f, den = 0.f;
    #pragma unroll
    for (int w = 0; w < 4; ++w) {
      float sc = __expf(mS[w][r] - M);
      num += sc * accS[w][r][d];
      den += sc * lS[w][r];
    }
    int b = qbase + r;
    int col = head * HDIM + d;
    o2[(size_t)b * 256 + col] = gbuf[(size_t)b * 256 + col] * (num * __builtin_amdgcn_rcpf(den));
  }
}

// ---------------------------------------------------------------------------
// Host orchestration
// ---------------------------------------------------------------------------
extern "C" void kernel_launch(void* const* d_in, const int* in_sizes, int n_in,
                              void* d_out, int out_size, void* d_ws, size_t ws_size,
                              hipStream_t stream) {
  const float* q    = (const float*)d_in[0];
  const float* k    = (const float*)d_in[1];
  const float* v    = (const float*)d_in[2];
  const float* q_sw = (const float*)d_in[3];
  const float* q_bw = (const float*)d_in[4];
  const float* q_bb = (const float*)d_in[5];
  const float* k_sw = (const float*)d_in[6];
  const float* k_bw = (const float*)d_in[7];
  const float* k_bb = (const float*)d_in[8];
  const float* v_sw = (const float*)d_in[9];
  const float* v_bw = (const float*)d_in[10];
  const float* v_bb = (const float*)d_in[11];
  const float* g_sw = (const float*)d_in[12];
  const float* g_bw = (const float*)d_in[13];
  const float* g_bb = (const float*)d_in[14];
  const float* o_sw = (const float*)d_in[15];
  const float* o_bw = (const float*)d_in[16];
  const float* o_bb = (const float*)d_in[17];

  // Workspace carve-up (~8.7 MB)
  unsigned short* wk16 = (unsigned short*)d_ws;
  unsigned short* wv16 = wk16 + (size_t)S_ROWS * 256;
  unsigned short* wq16 = wv16 + (size_t)S_ROWS * 256;
  float* gbuf = (float*)(wq16 + (size_t)B_ROWS * 256);
  float* o2   = gbuf + (size_t)B_ROWS * 256;

  dim3 blk(128);
  // wk = fastkan(k)  -> bf16                               (dominant GEMM)
  fastkan_kernel<<<dim3(S_ROWS / 16, 4), blk, 0, stream>>>(k, k_sw, k_bw, k_bb, 1.0f, 0, nullptr, wk16);
  // wv = fastkan(v)  -> bf16                               (dominant GEMM)
  fastkan_kernel<<<dim3(S_ROWS / 16, 4), blk, 0, stream>>>(v, v_sw, v_bw, v_bb, 1.0f, 0, nullptr, wv16);
  // wq = fastkan(q) * 1/sqrt(d) -> bf16
  fastkan_kernel<<<dim3(B_ROWS / 16, 4), blk, 0, stream>>>(q, q_sw, q_bw, q_bb, NORM, 0, nullptr, wq16);
  // g  = sigmoid(fastkan(q)) -> f32
  fastkan_kernel<<<dim3(B_ROWS / 16, 4), blk, 0, stream>>>(q, g_sw, g_bw, g_bb, 1.0f, 1, gbuf, nullptr);
  // o2 = g * softmax(wq @ wk^T) @ wv
  attention_kernel<<<dim3(HEADS, B_ROWS / 16), blk, 0, stream>>>(wq16, wk16, wv16, gbuf, o2);
  // out = fastkan(o2) -> f32 (d_out)
  fastkan_kernel<<<dim3(B_ROWS / 16, 4), blk, 0, stream>>>(o2, o_sw, o_bw, o_bb, 1.0f, 0, (float*)d_out, nullptr);
}
